// GIPPOBranch_21543555956855
// MI455X (gfx1250) — matrix operand order repaired and checked by reading, not compiled, not run on hardware
//
#include <hip/hip_runtime.h>
#include <hip/hip_bf16.h>

typedef __attribute__((ext_vector_type(16))) _Float16 v16h;
typedef __attribute__((ext_vector_type(8)))  _Float16 v8h;
typedef __attribute__((ext_vector_type(4)))  _Float16 v4h;
typedef __attribute__((ext_vector_type(8)))  float    v8f;

#define BATCH 1024
#define AGENTS 32
#define NN (BATCH * AGENTS)         // 32768 nodes
#define INF_ 18
#define EFEAT 5
#define OUTF 4
#define EPG (AGENTS * (AGENTS - 1)) // 992 edges per graph
#define DEG (AGENTS - 1)            // 31 in-edges per destination
#define ASTR 136                    // activation strip stride (halves), bank-conflict-free

// ---------------------------------------------------------------------------
// WMMA 16x16x32 f16 step with the B fragment ALREADY IN REGISTERS.
// A (16x32) read from LDS row-major, stride `as`: two contiguous 8-half runs
// per lane -> 2x ds_load_b128 (the only hot-loop LDS loads).
// ---------------------------------------------------------------------------
__device__ inline v8f wmma_step_r(const _Float16* aBase, int as, v16h b, v8f c) {
  int lane = threadIdx.x & 31;
  const _Float16* ap = aBase + (lane & 15) * as + ((lane >> 4) << 3);
  v16h a;
#pragma unroll
  for (int j = 0; j < 8; ++j) {
    a[j]     = ap[j];        // K = kh*8 + j
    a[8 + j] = ap[16 + j];   // K = 16 + kh*8 + j
  }
  return __builtin_amdgcn_wmma_f32_16x16x32_f16(false, a, false, b,
                                                (short)0, c, false, false);
}

// Load this lane's B fragment (16 contiguous halves, 32B aligned) from a
// pre-swizzled 512-half LDS tile into registers.
__device__ inline v16h load_bfrag(const _Float16* tile) {
  return *(const v16h*)(tile + (threadIdx.x & 31) * 16);
}

// ---------------------------------------------------------------------------
// Stage (Ka x Nc) fp32 weights (row offset krow0, leading dim ldw) into LDS
// f16 PRE-SWIZZLED into B-fragment tiles: tile (ki,ni) occupies 512 halves at
// dst + (ki*(Npad/16)+ni)*512; lane l holds elements [l*16..l*16+15], element
// j=2r+odd -> K = ki*32+(r>>2)*16+(l>>4)*8+(r&3)*2+odd, N = ni*16+(l&15).
// ---------------------------------------------------------------------------
__device__ inline void stage_weight_sw(const float* W, int Ka, int Nc, int ldw,
                                       _Float16* dst, int Kpad, int Npad, int krow0) {
  int ntiles = Npad >> 4;
  int total = (Kpad >> 5) * ntiles * 512;
  for (int idx = threadIdx.x; idx < total; idx += blockDim.x) {
    int tile = idx >> 9;
    int within = idx & 511;
    int l = within >> 4, j = within & 15;
    int r = j >> 1, odd = j & 1;
    int kh = l >> 4, nl = l & 15;
    int ki = tile / ntiles, ni = tile - ki * ntiles;
    int k = (ki << 5) + ((r >> 2) << 4) + (kh << 3) + ((r & 3) << 1) + odd;
    int n = (ni << 4) + nl;
    float v = 0.0f;
    if (k < Ka && n < Nc) v = W[(krow0 + k) * ldw + n];
    dst[idx] = (_Float16)v;
  }
}
__device__ inline void stage_bias(const float* b, int Nc, float* dst, int Npad) {
  for (int n = threadIdx.x; n < Npad; n += blockDim.x)
    dst[n] = (n < Nc) ? b[n] : 0.0f;
}

// Epilogue: C fragment + per-lane bias (in register), optional relu, f16 -> LDS.
__device__ inline void epilogue_lds(v8f c, float biasv, bool relu,
                                    _Float16* sOut, int n) {
  int kh = (threadIdx.x & 31) >> 4;
  _Float16* p = sOut + (kh << 3) * ASTR + n;
#pragma unroll
  for (int r = 0; r < 8; ++r) {
    float v = c[r] + biasv;
    if (relu) v = fmaxf(v, 0.0f);
    p[r * ASTR] = (_Float16)v;
  }
}

// ---------------------------------------------------------------------------
// Node kernel 1: enc MLP (18->32 relu ->64 relu ->64), then s = enc3@W1a + b1
// (W1a = first 64 rows of msg_enc W1). Output s: [N,128] f16.
// All B fragments + biases hoisted to registers; hot loop = A-loads only.
// ---------------------------------------------------------------------------
__global__ void node_enc_s_kernel(const float* __restrict__ obs,
                                  const float* eW1, const float* eB1,
                                  const float* eW2, const float* eB2,
                                  const float* eW3, const float* eB3,
                                  const float* msgW1, const float* msgB1,
                                  _Float16* __restrict__ sOutG) {
  __shared__ __align__(32) _Float16 sW1[32 * 32], sW2[32 * 64], sW3[64 * 64], sWa[64 * 128];
  __shared__ float sB1[32], sB2[64], sB3[64], sBa[128];
  __shared__ __align__(32) _Float16 sX[16 * ASTR], sY[16 * ASTR];

  stage_weight_sw(eW1, INF_, 32, 32, sW1, 32, 32, 0);
  stage_weight_sw(eW2, 32, 64, 64, sW2, 32, 64, 0);
  stage_weight_sw(eW3, 64, 64, 64, sW3, 64, 64, 0);
  stage_weight_sw(msgW1, 64, 128, 128, sWa, 64, 128, 0);
  stage_bias(eB1, 32, sB1, 32);
  stage_bias(eB2, 64, sB2, 64);
  stage_bias(eB3, 64, sB3, 64);
  stage_bias(msgB1, 128, sBa, 128);
  __syncthreads();

  int w = threadIdx.x >> 5, lane = threadIdx.x & 31;
  int kh = lane >> 4;
  int n = (w << 4) + (lane & 15);
  v16h f1 = {}, f2 = {}, f3a = {}, f3b = {};
  float b1v = 0.0f, b2v = 0.0f, b3v = 0.0f;
  if (w < 2) { f1 = load_bfrag(sW1 + w * 512); b1v = sB1[n]; }
  if (w < 4) {
    f2 = load_bfrag(sW2 + w * 512); b2v = sB2[n];
    f3a = load_bfrag(sW3 + (0 * 4 + w) * 512);
    f3b = load_bfrag(sW3 + (1 * 4 + w) * 512); b3v = sB3[n];
  }
  v16h fa0 = load_bfrag(sWa + (0 * 8 + w) * 512);
  v16h fa1 = load_bfrag(sWa + (1 * 8 + w) * 512);
  float bav = sBa[n];

  int strips = NN / 16;
  for (int strip = blockIdx.x; strip < strips; strip += gridDim.x) {
    for (int idx = threadIdx.x; idx < 16 * 32; idx += blockDim.x) {
      int m = idx >> 5, k = idx & 31;
      sX[m * ASTR + k] = (_Float16)((k < INF_) ? obs[(strip * 16 + m) * INF_ + k] : 0.0f);
    }
    __syncthreads();
    if (w < 2) { v8f c = {}; c = wmma_step_r(sX, ASTR, f1, c); epilogue_lds(c, b1v, true, sY, n); }
    __syncthreads();
    if (w < 4) { v8f c = {}; c = wmma_step_r(sY, ASTR, f2, c); epilogue_lds(c, b2v, true, sX, n); }
    __syncthreads();
    if (w < 4) {
      v8f c = {};
      c = wmma_step_r(sX, ASTR, f3a, c);
      c = wmma_step_r(sX + 32, ASTR, f3b, c);
      epilogue_lds(c, b3v, false, sY, n);
    }
    __syncthreads();
    {
      v8f c = {};
      c = wmma_step_r(sY, ASTR, fa0, c);
      c = wmma_step_r(sY + 32, ASTR, fa1, c);
      _Float16* p = sOutG + (strip * 16 + (kh << 3)) * 128 + n;  // const offsets
#pragma unroll
      for (int r = 0; r < 8; ++r) p[r * 128] = (_Float16)(c[r] + bav);
    }
    __syncthreads();
  }
}

// ---------------------------------------------------------------------------
// Node kernel 2: loc MLP (18->64 relu ->128 relu ->128) -> local_enc [N,128] f16
// ---------------------------------------------------------------------------
__global__ void node_loc_kernel(const float* __restrict__ obs,
                                const float* lW1, const float* lB1,
                                const float* lW2, const float* lB2,
                                const float* lW3, const float* lB3,
                                _Float16* __restrict__ localG) {
  __shared__ __align__(32) _Float16 sW1[32 * 64], sW2[64 * 128], sW3[128 * 128];
  __shared__ float sB1[64], sB2[128], sB3[128];
  __shared__ __align__(32) _Float16 sX[16 * ASTR], sY[16 * ASTR];

  stage_weight_sw(lW1, INF_, 64, 64, sW1, 32, 64, 0);
  stage_weight_sw(lW2, 64, 128, 128, sW2, 64, 128, 0);
  stage_weight_sw(lW3, 128, 128, 128, sW3, 128, 128, 0);
  stage_bias(lB1, 64, sB1, 64);
  stage_bias(lB2, 128, sB2, 128);
  stage_bias(lB3, 128, sB3, 128);
  __syncthreads();

  int w = threadIdx.x >> 5, lane = threadIdx.x & 31;
  int kh = lane >> 4;
  int n = (w << 4) + (lane & 15);
  v16h f1 = {};
  float b1v = 0.0f;
  if (w < 4) { f1 = load_bfrag(sW1 + w * 512); b1v = sB1[n]; }
  v16h f20 = load_bfrag(sW2 + (0 * 8 + w) * 512);
  v16h f21 = load_bfrag(sW2 + (1 * 8 + w) * 512);
  v16h f30 = load_bfrag(sW3 + (0 * 8 + w) * 512);
  v16h f31 = load_bfrag(sW3 + (1 * 8 + w) * 512);
  v16h f32_ = load_bfrag(sW3 + (2 * 8 + w) * 512);
  v16h f33 = load_bfrag(sW3 + (3 * 8 + w) * 512);
  float b2v = sB2[n], b3v = sB3[n];

  int strips = NN / 16;
  for (int strip = blockIdx.x; strip < strips; strip += gridDim.x) {
    for (int idx = threadIdx.x; idx < 16 * 32; idx += blockDim.x) {
      int m = idx >> 5, k = idx & 31;
      sX[m * ASTR + k] = (_Float16)((k < INF_) ? obs[(strip * 16 + m) * INF_ + k] : 0.0f);
    }
    __syncthreads();
    if (w < 4) { v8f c = {}; c = wmma_step_r(sX, ASTR, f1, c); epilogue_lds(c, b1v, true, sY, n); }
    __syncthreads();
    {
      v8f c = {};
      c = wmma_step_r(sY, ASTR, f20, c);
      c = wmma_step_r(sY + 32, ASTR, f21, c);
      epilogue_lds(c, b2v, true, sX, n);
    }
    __syncthreads();
    {
      v8f c = {};
      c = wmma_step_r(sX, ASTR, f30, c);
      c = wmma_step_r(sX + 32, ASTR, f31, c);
      c = wmma_step_r(sX + 64, ASTR, f32_, c);
      c = wmma_step_r(sX + 96, ASTR, f33, c);
      _Float16* p = localG + (strip * 16 + (kh << 3)) * 128 + n;  // const offsets
#pragma unroll
      for (int r = 0; r < 8; ++r) p[r * 128] = (_Float16)(c[r] + b3v);
    }
    __syncthreads();
  }
}

// ---------------------------------------------------------------------------
// Fused edge kernel, destination-major: one strip = ONE destination node's 31
// in-edges (+1 zero-pad row) = a 32-row / 2-M-tile strip. Pipeline per strip:
//   edge_attr(5) -> edge_enc (5->32 relu ->32)
//   h1  = relu(ea @ W1b + s[src])   (s is f16; b1 folded into s; pad row -> 0)
//   msg = h1 @ W2 ; column-sum over 32 rows (register partials + one shfl_xor
//   cross-half combine) ; agg[dst] = sum + 31*b2  -- plain f16 store,
//   no atomics, no agg zero-init. B fragments reused across both M-tiles.
// ---------------------------------------------------------------------------
__global__ void edge_kernel(const float* __restrict__ pos,
                            const float* __restrict__ vel,
                            const int* __restrict__ ei,
                            const float* egW1, const float* egB1,
                            const float* egW2, const float* egB2,
                            const float* msgW1,
                            const float* msgW2, const float* msgB2,
                            const _Float16* __restrict__ sG,
                            _Float16* __restrict__ aggG) {
  __shared__ __align__(32) _Float16 sWe1[32 * 32], sWe2[32 * 32], sW1b[32 * 128], sW2[128 * 128];
  __shared__ float sBe1[32], sBe2[32], sB2[128];
  __shared__ __align__(32) _Float16 sX[32 * ASTR], sY[32 * ASTR];
  __shared__ int sOff[32];   // per-row byte offset into s (src*256)

  stage_weight_sw(egW1, EFEAT, 32, 32, sWe1, 32, 32, 0);
  stage_weight_sw(egW2, 32, 32, 32, sWe2, 32, 32, 0);
  stage_weight_sw(msgW1, 32, 128, 128, sW1b, 32, 128, 64);  // rows 64..95
  stage_weight_sw(msgW2, 128, 128, 128, sW2, 128, 128, 0);
  stage_bias(egB1, 32, sBe1, 32);
  stage_bias(egB2, 32, sBe2, 32);
  stage_bias(msgB2, 128, sB2, 128);
  __syncthreads();

  int w = threadIdx.x >> 5, lane = threadIdx.x & 31;
  int kh = lane >> 4;
  int n = (w << 4) + (lane & 15);
  v16h fe1 = {}, fe2 = {};
  float be1v = 0.0f, be2v = 0.0f;
  if (w < 2) {
    fe1 = load_bfrag(sWe1 + w * 512); be1v = sBe1[n];
    fe2 = load_bfrag(sWe2 + w * 512); be2v = sBe2[n];
  }
  v16h f1b = load_bfrag(sW1b + w * 512);
  v16h f20 = load_bfrag(sW2 + (0 * 8 + w) * 512);
  v16h f21 = load_bfrag(sW2 + (1 * 8 + w) * 512);
  v16h f22 = load_bfrag(sW2 + (2 * 8 + w) * 512);
  v16h f23 = load_bfrag(sW2 + (3 * 8 + w) * 512);
  float b2v = sB2[n];
  const char* sBase = (const char*)sG + n * 2;  // uniform 64-bit base + n

  // strip == global destination node index (b*32 + i)
  for (int dst = blockIdx.x; dst < NN; dst += gridDim.x) {
    int b = dst >> 5;          // graph index
    int i = dst & 31;          // agent (target) index within graph
    if (threadIdx.x < 32) {
      int t = threadIdx.x;
      _Float16* row = sX + t * ASTR;
      if (t < DEG) {
        int le = i * DEG + t;               // edges for dst i are contiguous
        int src = ei[le] + b * AGENTS;
        sOff[t] = src * 256;                // byte offset of s row
        float rpx = pos[src * 2 + 0] - pos[dst * 2 + 0];
        float rpy = pos[src * 2 + 1] - pos[dst * 2 + 1];
        float d = sqrtf(rpx * rpx + rpy * rpy);
        float rvx = vel[src * 2 + 0] - vel[dst * 2 + 0];
        float rvy = vel[src * 2 + 1] - vel[dst * 2 + 1];
        row[0] = (_Float16)rpx; row[1] = (_Float16)rpy; row[2] = (_Float16)d;
        row[3] = (_Float16)rvx; row[4] = (_Float16)rvy;
        for (int k = EFEAT; k < 32; ++k) row[k] = (_Float16)0.0f;
      } else {                               // pad row
        sOff[t] = 0;
        for (int k = 0; k < 32; ++k) row[k] = (_Float16)0.0f;
      }
    }
    __syncthreads();
    if (w < 2) {  // edge hidden (5->32 relu), both M-tiles share the B frag
      v8f c0 = {}, c1 = {};
      c0 = wmma_step_r(sX, ASTR, fe1, c0);
      c1 = wmma_step_r(sX + 16 * ASTR, ASTR, fe1, c1);
      epilogue_lds(c0, be1v, true, sY, n);
      epilogue_lds(c1, be1v, true, sY + 16 * ASTR, n);
    }
    __syncthreads();
    if (w < 2) {  // ea (32->32)
      v8f c0 = {}, c1 = {};
      c0 = wmma_step_r(sY, ASTR, fe2, c0);
      c1 = wmma_step_r(sY + 16 * ASTR, ASTR, fe2, c1);
      epilogue_lds(c0, be2v, false, sX, n);
      epilogue_lds(c1, be2v, false, sX + 16 * ASTR, n);
    }
    __syncthreads();
    // h1 = relu(ea @ W1b + s[src]) ; pad row forced to 0
    {
      v8f c0 = {}, c1 = {};
      c0 = wmma_step_r(sX, ASTR, f1b, c0);
      c1 = wmma_step_r(sX + 16 * ASTR, ASTR, f1b, c1);
      _Float16* p0 = sY + (kh << 3) * ASTR + n;
#pragma unroll
      for (int r = 0; r < 8; ++r) {
        int m = (kh << 3) + r;
        float g0 = (float)*(const _Float16*)(sBase + sOff[m]);
        p0[r * ASTR] = (_Float16)fmaxf(c0[r] + g0, 0.0f);
        int m2 = 16 + m;
        float g1 = (float)*(const _Float16*)(sBase + sOff[m2]);
        float v1 = (m2 == DEG) ? 0.0f : fmaxf(c1[r] + g1, 0.0f);
        p0[(16 + r) * ASTR] = (_Float16)v1;
      }
      __syncthreads();
    }
    // msg = h1 @ W2 ; column-sum over 32 rows ; agg[dst] = sum + 31*b2
    {
      v8f c0 = {}, c1 = {};
      c0 = wmma_step_r(sY, ASTR, f20, c0);
      c1 = wmma_step_r(sY + 16 * ASTR, ASTR, f20, c1);
      c0 = wmma_step_r(sY + 32, ASTR, f21, c0);
      c1 = wmma_step_r(sY + 16 * ASTR + 32, ASTR, f21, c1);
      c0 = wmma_step_r(sY + 64, ASTR, f22, c0);
      c1 = wmma_step_r(sY + 16 * ASTR + 64, ASTR, f22, c1);
      c0 = wmma_step_r(sY + 96, ASTR, f23, c0);
      c1 = wmma_step_r(sY + 16 * ASTR + 96, ASTR, f23, c1);
      float part = 0.0f;
#pragma unroll
      for (int r = 0; r < 8; ++r) part += c0[r] + c1[r];
      part += __shfl_xor(part, 16, 32);   // combine the two row-halves
      if (kh == 0)
        aggG[dst * 128 + n] = (_Float16)(part + (float)DEG * b2v);
      __syncthreads();
    }
  }
}

// ---------------------------------------------------------------------------
// Generic single-layer GEMM on f16 activations, fully constant-folded via
// template parameters (lda == KA, ldc == NC, ldadd == 128 at every call):
//   C = act(A[M,KA] @ W[KA,NC] + b (+addend)), A/addend f16, C f16 or f32.
// A staging is a pure 16B vector copy; store epilogue is one base pointer +
// constant-offset clause. Up to 4 register weight fragments per wave.
// ---------------------------------------------------------------------------
template <int KA, int NC, int RELU, int F32OUT>
__global__ void gemm_strip_kernel(const _Float16* __restrict__ Amat,
                                  const float* __restrict__ W,
                                  const float* __restrict__ bias,
                                  const _Float16* __restrict__ addend,
                                  void* __restrict__ Cmat, int M) {
  constexpr int NPAD = (NC + 15) & ~15;
  constexpr int NT = NPAD >> 4;
  constexpr int NK = KA >> 5;
  constexpr int KS = KA + 8;
  constexpr int CH = KA >> 3;   // 16B chunks per row
  __shared__ __align__(32) _Float16 sW[NK * NT * 512];
  __shared__ __align__(32) _Float16 sA[16 * KS];
  __shared__ float sB[NPAD];
  stage_weight_sw(W, KA, NC, NC, sW, KA, NPAD, 0);
  stage_bias(bias, NC, sB, NPAD);
  __syncthreads();

  int w = threadIdx.x >> 5, lane = threadIdx.x & 31;
  int kh = lane >> 4;
  int n = (w << 4) + (lane & 15);
  v16h wf[NK];
  float bv = 0.0f;
  if (w < NT) {
#pragma unroll
    for (int ki = 0; ki < NK; ++ki)
      wf[ki] = load_bfrag(sW + (ki * NT + w) * 512);
    bv = sB[n];
  }

  int strips = M >> 4;
  for (int strip = blockIdx.x; strip < strips; strip += gridDim.x) {
    for (int idx = threadIdx.x; idx < 16 * CH; idx += blockDim.x) {
      int m = idx / CH, ch = idx - m * CH;
      *(v8h*)(sA + m * KS + ch * 8) =
          *(const v8h*)(Amat + (strip * 16 + m) * KA + ch * 8);
    }
    __syncthreads();
    if (w < NT) {
      v8f c = {};
#pragma unroll
      for (int ki = 0; ki < NK; ++ki)
        c = wmma_step_r(sA + ki * 32, KS, wf[ki], c);
      if (n < NC) {
        int row0 = strip * 16 + (kh << 3);
        float vr[8];
#pragma unroll
        for (int r = 0; r < 8; ++r) vr[r] = c[r] + bv;
        if (addend) {
          const _Float16* ap = addend + row0 * 128 + n;
#pragma unroll
          for (int r = 0; r < 8; ++r) vr[r] += (float)ap[r * 128];
        }
        if (RELU) {
#pragma unroll
          for (int r = 0; r < 8; ++r) vr[r] = fmaxf(vr[r], 0.0f);
        }
        if (F32OUT) {
          float* pf = (float*)Cmat + row0 * NC + n;
#pragma unroll
          for (int r = 0; r < 8; ++r) pf[r * NC] = vr[r];
        } else {
          _Float16* ph = (_Float16*)Cmat + row0 * NC + n;
#pragma unroll
          for (int r = 0; r < 8; ++r) ph[r * NC] = (_Float16)vr[r];
        }
      }
    }
    __syncthreads();
  }
}

// ---------------------------------------------------------------------------
// LayerNorm over 128 features, one wave per row (wave32 shuffle reduction),
// f16 in/out with 8B vector accesses.
// ---------------------------------------------------------------------------
__global__ void ln_kernel(const _Float16* __restrict__ in,
                          const float* __restrict__ g,
                          const float* __restrict__ bt,
                          _Float16* __restrict__ out, int M) {
  int row = blockIdx.x * (blockDim.x >> 5) + (threadIdx.x >> 5);
  if (row >= M) return;
  int lane = threadIdx.x & 31;
  v4h t4 = ((const v4h*)(in + row * 128))[lane];
  float4 gv = ((const float4*)g)[lane];
  float4 bv = ((const float4*)bt)[lane];
  float tx = (float)t4[0], ty = (float)t4[1], tz = (float)t4[2], tw = (float)t4[3];
  float sum = tx + ty + tz + tw;
  for (int off = 16; off > 0; off >>= 1) sum += __shfl_xor(sum, off, 32);
  float mu = sum * (1.0f / 128.0f);
  float dx = tx - mu, dy = ty - mu, dz = tz - mu, dw = tw - mu;
  float var = dx * dx + dy * dy + dz * dz + dw * dw;
  for (int off = 16; off > 0; off >>= 1) var += __shfl_xor(var, off, 32);
  float r = rsqrtf(var * (1.0f / 128.0f) + 1e-5f);
  v4h o;
  o[0] = (_Float16)(dx * r * gv.x + bv.x);
  o[1] = (_Float16)(dy * r * gv.y + bv.y);
  o[2] = (_Float16)(dz * r * gv.z + bv.z);
  o[3] = (_Float16)(dw * r * gv.w + bv.w);
  ((v4h*)(out + row * 128))[lane] = o;
}

// ---------------------------------------------------------------------------
// Launch. Input order = jax pytree flatten of setup_inputs() (dicts sorted by
// key): obs,pos,vel,edge_index, then params: dec,edge_enc,enc,head,head_ln,
// lin_agg,loc,msg_enc; each MLP as W1,b1,W2,b2,...
// All intermediate buffers are f16 (halves HBM/L2 traffic); d_out is fp32.
// ---------------------------------------------------------------------------
extern "C" void kernel_launch(void* const* d_in, const int* in_sizes, int n_in,
                              void* d_out, int out_size, void* d_ws, size_t ws_size,
                              hipStream_t stream) {
  const float* obs = (const float*)d_in[0];
  const float* pos = (const float*)d_in[1];
  const float* vel = (const float*)d_in[2];
  const int*   ei  = (const int*)d_in[3];
  const float *decW1 = (const float*)d_in[4],  *decB1 = (const float*)d_in[5];
  const float *decW2 = (const float*)d_in[6],  *decB2 = (const float*)d_in[7];
  const float *decW3 = (const float*)d_in[8],  *decB3 = (const float*)d_in[9];
  const float *egW1  = (const float*)d_in[10], *egB1  = (const float*)d_in[11];
  const float *egW2  = (const float*)d_in[12], *egB2  = (const float*)d_in[13];
  const float *encW1 = (const float*)d_in[14], *encB1 = (const float*)d_in[15];
  const float *encW2 = (const float*)d_in[16], *encB2 = (const float*)d_in[17];
  const float *encW3 = (const float*)d_in[18], *encB3 = (const float*)d_in[19];
  const float *hW1   = (const float*)d_in[20], *hB1   = (const float*)d_in[21];
  const float *hW2   = (const float*)d_in[22], *hB2   = (const float*)d_in[23];
  const float *hW3   = (const float*)d_in[24], *hB3   = (const float*)d_in[25];
  const float *lnG   = (const float*)d_in[26], *lnB   = (const float*)d_in[27];
  const float *linW  = (const float*)d_in[28], *linB  = (const float*)d_in[29];
  const float *locW1 = (const float*)d_in[30], *locB1 = (const float*)d_in[31];
  const float *locW2 = (const float*)d_in[32], *locB2 = (const float*)d_in[33];
  const float *locW3 = (const float*)d_in[34], *locB3 = (const float*)d_in[35];
  const float *msgW1 = (const float*)d_in[36], *msgB1 = (const float*)d_in[37];
  const float *msgW2 = (const float*)d_in[38], *msgB2 = (const float*)d_in[39];

  _Float16* buf0 = (_Float16*)d_ws;             // s,         then reused
  _Float16* buf1 = buf0 + (size_t)NN * 128;     // local_enc, then reused
  _Float16* buf2 = buf1 + (size_t)NN * 128;     // agg,       then reused

  node_enc_s_kernel<<<2048, 256, 0, stream>>>(obs, encW1, encB1, encW2, encB2,
                                              encW3, encB3, msgW1, msgB1, buf0);
  node_loc_kernel<<<2048, 256, 0, stream>>>(obs, locW1, locB1, locW2, locB2,
                                            locW3, locB3, buf1);
  // destination-major: one strip per destination node, plain stores into agg
  edge_kernel<<<4096, 256, 0, stream>>>(pos, vel, ei, egW1, egB1, egW2, egB2,
                                        msgW1, msgW2, msgB2, buf0, buf2);

  // gnn = agg @ Wl + bl + local_enc          (buf2,buf1 -> buf0)
  gemm_strip_kernel<128, 128, 0, 0><<<2048, 256, 0, stream>>>(buf2, linW, linB, buf1, buf0, NN);
  // dec: 128->128 relu ->128 relu ->128
  gemm_strip_kernel<128, 128, 1, 0><<<2048, 256, 0, stream>>>(buf0, decW1, decB1, nullptr, buf1, NN);
  gemm_strip_kernel<128, 128, 1, 0><<<2048, 256, 0, stream>>>(buf1, decW2, decB2, nullptr, buf2, NN);
  gemm_strip_kernel<128, 128, 0, 0><<<2048, 256, 0, stream>>>(buf2, decW3, decB3, nullptr, buf0, NN);
  // layernorm
  ln_kernel<<<NN / 8, 256, 0, stream>>>(buf0, lnG, lnB, buf1, NN);
  // head: 128->64 relu ->32 relu ->4 (final layer emits fp32 to d_out)
  gemm_strip_kernel<128, 64, 1, 0><<<2048, 256, 0, stream>>>(buf1, hW1, hB1, nullptr, buf2, NN);
  gemm_strip_kernel<64, 32, 1, 0><<<2048, 256, 0, stream>>>(buf2, hW2, hB2, nullptr, buf0, NN);
  gemm_strip_kernel<32, 4, 0, 1><<<2048, 256, 0, stream>>>(buf0, hW3, hB3, nullptr, d_out, NN);
}
